// cross_channel_interp_91130616087186
// MI455X (gfx1250) — matrix operand order repaired and checked by reading, not compiled, not run on hardware
//
#include <hip/hip_runtime.h>
#include <hip/hip_bf16.h>

// Problem constants (match reference)
constexpr int kB  = 64;
constexpr int kD  = 512;
constexpr int kT  = 1024;
constexpr int kC3 = 3 * kD;              // 1536 channels

// GEMM tiling
constexpr int kTileT = 128;              // t per workgroup
constexpr int kEChunk = 64;              // e per LDS M-tile
constexpr int kPadK  = 520;              // 512 K + 8 bf16 pad (16B) per LDS row

// LDS layout (dynamic) for the GEMM kernel; dynamic LDS assumed to start at 0
constexpr unsigned kAtBytes  = kTileT * kPadK * 2;     // 133120: aT[t][d]
constexpr unsigned kMtBytes  = kEChunk * kPadK * 2;    //  66560: one mT buffer
constexpr unsigned kStatsOff = kAtBytes + 2 * kMtBytes;
constexpr size_t   kSmemK3   = kStatsOff + (kD + kTileT + kTileT) * sizeof(float);

typedef __attribute__((ext_vector_type(16))) __bf16   v16bf;
typedef __attribute__((ext_vector_type(8)))  __bf16   bf16x8;
typedef __attribute__((ext_vector_type(8)))  float    v8f;
typedef __attribute__((ext_vector_type(4)))  unsigned u32x4;
typedef __attribute__((ext_vector_type(8)))  int      i32x8;
typedef __attribute__((ext_vector_type(4)))  int      i32x4;

#define BF16CAT(x, y) __builtin_shufflevector((x), (y), 0,1,2,3,4,5,6,7,8,9,10,11,12,13,14,15)

static __device__ inline unsigned pack2bf(float a, float b) {
  const unsigned short ua = __builtin_bit_cast(unsigned short, (__bf16)a);
  const unsigned short ub = __builtin_bit_cast(unsigned short, (__bf16)b);
  return (unsigned)ua | ((unsigned)ub << 16);
}

// ---------------------------------------------------------------------------
// Kernel 0: Mt[e][d] = (bf16) M[d][e]  — one-time transpose+convert, 512 KB,
// stays resident in the 192 MB L2 for all GEMM workgroups / TDM reads.
// ---------------------------------------------------------------------------
__global__ void cci_mprep_kernel(const float* __restrict__ M,
                                 __bf16* __restrict__ Mt) {
  __shared__ float tile[32][33];
  const int e0 = blockIdx.x * 32, d0 = blockIdx.y * 32;
  const int tx = threadIdx.x & 31, ty = threadIdx.x >> 5;   // 32 x 8
  #pragma unroll
  for (int j = 0; j < 32; j += 8)
    tile[ty + j][tx] = M[(size_t)(d0 + ty + j) * kD + e0 + tx];
  __syncthreads();
  #pragma unroll
  for (int j = 0; j < 32; j += 8)
    Mt[(size_t)(e0 + ty + j) * kD + d0 + tx] = (__bf16)tile[tx][ty + j];
}

// ---------------------------------------------------------------------------
// Kernel 1: mean[b,d] = (1/T) sum_t x[b,d,t]
// ---------------------------------------------------------------------------
__global__ void cci_mean_kernel(const float* __restrict__ x,
                                float* __restrict__ mean) {
  __shared__ float red[256];
  const int row = blockIdx.x;            // 0 .. B*D-1
  const int b = row >> 9;
  const int d = row & (kD - 1);
  const int tid = threadIdx.x;
  const float4 v = ((const float4*)&x[((size_t)b * kC3 + d) * kT])[tid];
  red[tid] = v.x + v.y + v.z + v.w;
  __syncthreads();
  for (int s = 128; s > 0; s >>= 1) {
    if (tid < s) red[tid] += red[tid + s];
    __syncthreads();
  }
  if (tid == 0) mean[row] = red[0] * (1.0f / (float)kT);
}

// ---------------------------------------------------------------------------
// Kernel 2: per-(b,t) online softmax stats over channels (coalesced across t),
// fused with intensity output out[b,D+d,t] = exp(w).
// ---------------------------------------------------------------------------
__global__ void cci_softmax_stats_kernel(const float* __restrict__ x,
                                         float* __restrict__ out,
                                         float* __restrict__ rowmax,
                                         float* __restrict__ rowinv) {
  const int t = blockIdx.x * 256 + threadIdx.x;
  const int b = blockIdx.y;
  const size_t base = (size_t)b * kC3 * kT + (size_t)kD * kT + t;
  float m = -3.402823466e38f;
  float s = 0.0f;
  for (int d = 0; d < kD; ++d) {
    const float v = x[base + (size_t)d * kT];
    out[base + (size_t)d * kT] = __expf(v);      // intensity
    const float mn = fmaxf(m, v);
    s = s * __expf(m - mn) + __expf(v - mn);
    m = mn;
  }
  rowmax[b * kT + t] = m;
  rowinv[b * kT + t] = 1.0f / s;
}

// ---------------------------------------------------------------------------
// TDM: DMA one 64 x 512 bf16 M-tile (row stride 1024B) into LDS, inserting
// 16B (4 DWORDs) of padding after every 1024B (256 DWORDs) row so the tile
// lands directly in the kPadK-padded WMMA layout.
// ---------------------------------------------------------------------------
static __device__ inline void tdm_issue_mtile(unsigned lds_off,
                                              const __bf16* gsrc) {
  const unsigned long long ga = (unsigned long long)(uintptr_t)gsrc;
  u32x4 g0;
  g0[0] = 1u;                                           // count=1, user mode
  g0[1] = lds_off;                                      // lds_addr
  g0[2] = (unsigned)ga;                                 // global_addr[31:0]
  g0[3] = ((unsigned)(ga >> 32) & 0x1FFFFFFu)           // global_addr[56:32]
          | 0x80000000u;                                // type=2 ("image")
  i32x8 g1;
  g1[0] = (1 << 16)        // data_size = 2 bytes
        | (1 << 20)        // pad_enable
        | (7 << 22)        // pad_interval: 256 DWORDs (=1024B row)
        | (3 << 25);       // pad_amount: 4 DWORDs (=16B)
  g1[1] = (int)(512u << 16);   // tensor_dim0 = 512 (d, contiguous)
  g1[2] = (int)(512u << 16);   // tensor_dim1 = 512 (e rows)
  g1[3] = (int)(512u << 16);   // tile_dim0 = 512
  g1[4] = kEChunk;             // tile_dim1 = 64, tile_dim2 = 0
  g1[5] = 512;                 // tensor_dim0_stride = 512 elements
  g1[6] = 0;
  g1[7] = 0;
  const i32x4 z4 = {0, 0, 0, 0};
#if __clang_major__ >= 23
  const i32x8 z8 = {0, 0, 0, 0, 0, 0, 0, 0};
  __builtin_amdgcn_tensor_load_to_lds(g0, g1, z4, z4, z8, 0);
#else
  __builtin_amdgcn_tensor_load_to_lds(g0, g1, z4, z4, 0);
#endif
}

// Per-k-chunk WMMA fragments (2 e-frags x 2 t-frags per wave)
struct KFrags {
  bf16x8 a00, a01, a10, a11, b00, b01, b10, b11;
};

static __device__ inline KFrags load_kfrags(const __bf16* mA0, const __bf16* mA1,
                                            const __bf16* bT0, const __bf16* bT1,
                                            int kb, int kbA, int kbB) {
  KFrags f;
  f.a00 = *(const bf16x8*)(mA0 + kb + kbA);
  f.a01 = *(const bf16x8*)(mA0 + kb + kbA + 16);
  f.a10 = *(const bf16x8*)(mA1 + kb + kbA);
  f.a11 = *(const bf16x8*)(mA1 + kb + kbA + 16);
  f.b00 = *(const bf16x8*)(bT0 + kb + kbB);
  f.b01 = *(const bf16x8*)(bT0 + kb + kbB + 16);
  f.b10 = *(const bf16x8*)(bT1 + kb + kbB);
  f.b11 = *(const bf16x8*)(bT1 + kb + kbB + 16);
  return f;
}

// ---------------------------------------------------------------------------
// Kernel 3: fused bf16-WMMA GEMM  C[e,t] = sum_d Mt[e,d] * a[t,d],
//   a[t,d] = softmax(w)[b,t,d] * (y[b,t,d] - mean[b,d])   (built once in LDS),
// M-tiles DMAed by the TDM (double-buffered, overlapped with compute).
// Workgroup: 256 thr (8 waves = 2 e-waves x 4 t-waves), tile 128 t x 512 e.
// ---------------------------------------------------------------------------
__global__ void cci_wmma_gemm_kernel(const float* __restrict__ x,
                                     const __bf16* __restrict__ Mt,
                                     const float* __restrict__ mean,
                                     const float* __restrict__ rowmax,
                                     const float* __restrict__ rowinv,
                                     float* __restrict__ out) {
  extern __shared__ __align__(16) char smem[];
  __bf16* aT   = (__bf16*)smem;                          // [128][520]
  float* smean = (float*)(smem + kStatsOff);             // [512]
  float* smax  = smean + kD;                             // [128]
  float* sinv  = smax + kTileT;                          // [128]

  const int tid = threadIdx.x;
  const int b   = blockIdx.y;
  const int t0  = blockIdx.x * kTileT;
  const size_t xb = (size_t)b * kC3 * kT;

  const int wv = tid >> 5;
  const bool wave0 = (__builtin_amdgcn_readfirstlane(wv) == 0);  // scalar branch

  // Kick off the TDM for the first M-tile immediately (overlaps aT build).
  if (wave0) tdm_issue_mtile(kAtBytes, Mt);

  // Stage softmax stats + channel means.
  if (tid < kTileT)            smax[tid]          = rowmax[b * kT + t0 + tid];
  else                         sinv[tid - kTileT] = rowinv[b * kT + t0 + (tid - kTileT)];
  smean[tid]       = mean[b * kD + tid];
  smean[tid + 256] = mean[b * kD + tid + 256];
  __syncthreads();

  // Build aT[t][d] (bf16, packed-pair ds_store_b32), reading y/w slabs once.
  for (int it = 0; it < 32; ++it) {
    const int idx = it * 256 + tid;          // 0..8191
    const int d   = (idx >> 5) << 1;         // even d, 0..510
    const int t4  = (idx & 31) << 2;         // 0..124
    const float4 y0 = *(const float4*)&x[xb + (size_t)d * kT + t0 + t4];
    const float4 y1 = *(const float4*)&x[xb + (size_t)(d + 1) * kT + t0 + t4];
    const float4 w0 = *(const float4*)&x[xb + (size_t)(kD + d) * kT + t0 + t4];
    const float4 w1 = *(const float4*)&x[xb + (size_t)(kD + d + 1) * kT + t0 + t4];
    const float mu0 = smean[d], mu1 = smean[d + 1];
    #pragma unroll
    for (int j = 0; j < 4; ++j) {
      const int t = t4 + j;
      const float sw = sinv[t];
      const float mx = smax[t];
      const float yj0 = (j == 0) ? y0.x : (j == 1) ? y0.y : (j == 2) ? y0.z : y0.w;
      const float yj1 = (j == 0) ? y1.x : (j == 1) ? y1.y : (j == 2) ? y1.z : y1.w;
      const float wj0 = (j == 0) ? w0.x : (j == 1) ? w0.y : (j == 2) ? w0.z : w0.w;
      const float wj1 = (j == 0) ? w1.x : (j == 1) ? w1.y : (j == 2) ? w1.z : w1.w;
      const float a0 = __expf(wj0 - mx) * sw * (yj0 - mu0);
      const float a1 = __expf(wj1 - mx) * sw * (yj1 - mu1);
      *(unsigned*)&aT[t * kPadK + d] = pack2bf(a0, a1);
    }
  }
  if (wave0) __builtin_amdgcn_s_wait_tensorcnt(0);  // first M-tile landed
  __syncthreads();                                   // aT + mT(0) ready

  // Wave/lane fragment geometry (ISA 7.12.2: 16-bit A 16x32 / B 32x16, wave32)
  const int lane  = tid & 31;
  const int waveE = (wv & 1) * 32;        // 2 e-waves x 2 frags = 64 e
  const int waveT = (wv >> 1) * 32;       // 4 t-waves x 2 frags = 128 t
  const int hi    = lane >> 4;
  const int lo    = lane & 15;
  const int kbA   = hi * 8;               // A: lanes16-31 start K=8 (then +16)
  const int kbB   = hi * 8;               // B: lanes16-31 start K=8 (then +16)
  const __bf16* bT0r = aT + (waveT + lo) * kPadK;
  const __bf16* bT1r = aT + (waveT + 16 + lo) * kPadK;

  for (int ec = 0; ec < kD / kEChunk; ++ec) {
    // Prefetch next M-tile into the other buffer while we compute this one.
    if (wave0 && ec + 1 < kD / kEChunk)
      tdm_issue_mtile(kAtBytes + ((unsigned)(ec + 1) & 1u) * kMtBytes,
                      Mt + (size_t)(ec + 1) * kEChunk * kD);

    const __bf16* mTbuf = (const __bf16*)(smem + kAtBytes + ((unsigned)ec & 1u) * kMtBytes);
    const __bf16* mA0 = mTbuf + (waveE + lo) * kPadK;
    const __bf16* mA1 = mTbuf + (waveE + 16 + lo) * kPadK;

    v8f acc00 = {}, acc01 = {}, acc10 = {}, acc11 = {};
    KFrags f = load_kfrags(mA0, mA1, bT0r, bT1r, 0, kbA, kbB);
    #pragma unroll
    for (int kc = 0; kc < 16; ++kc) {
      KFrags nf = f;
      if (kc < 15)  // software pipeline: issue next loads before consuming
        nf = load_kfrags(mA0, mA1, bT0r, bT1r, (kc + 1) * 32, kbA, kbB);
      const v16bf A0 = BF16CAT(f.a00, f.a01);
      const v16bf A1 = BF16CAT(f.a10, f.a11);
      const v16bf B0 = BF16CAT(f.b00, f.b01);
      const v16bf B1 = BF16CAT(f.b10, f.b11);
      acc00 = __builtin_amdgcn_wmma_f32_16x16x32_bf16(false, A0, false, B0, (short)0, acc00, false, false);
      acc01 = __builtin_amdgcn_wmma_f32_16x16x32_bf16(false, A0, false, B1, (short)0, acc01, false, false);
      acc10 = __builtin_amdgcn_wmma_f32_16x16x32_bf16(false, A1, false, B0, (short)0, acc10, false, false);
      acc11 = __builtin_amdgcn_wmma_f32_16x16x32_bf16(false, A1, false, B1, (short)0, acc11, false, false);
      f = nf;
    }

    // Epilogue: C layout — vgpr r: lanes0-15 M=r, lanes16-31 M=8+r; N=lo.
    #pragma unroll
    for (int i = 0; i < 2; ++i) {
      #pragma unroll
      for (int j = 0; j < 2; ++j) {
        const v8f acc = (i == 0) ? (j == 0 ? acc00 : acc01)
                                 : (j == 0 ? acc10 : acc11);
        const int t_g = t0 + waveT + j * 16 + lo;
        #pragma unroll
        for (int r = 0; r < 8; ++r) {
          const int e_g = ec * kEChunk + waveE + i * 16 + hi * 8 + r;
          const float rep = acc[r] + smean[e_g];
          out[xb + (size_t)e_g * kT + t_g] = rep;                   // rep1
          const size_t o2 = xb + (size_t)(2 * kD + e_g) * kT + t_g;
          out[o2] = x[o2] - rep;                                    // y_trans
        }
      }
    }
    if (wave0) __builtin_amdgcn_s_wait_tensorcnt(0);  // next tile landed
    __syncthreads();  // everyone done with this buffer / next tile visible
  }
}

// ---------------------------------------------------------------------------
extern "C" void kernel_launch(void* const* d_in, const int* in_sizes, int n_in,
                              void* d_out, int out_size, void* d_ws, size_t ws_size,
                              hipStream_t stream) {
  const float* x = (const float*)d_in[0];      // (B, 3D, T) fp32
  const float* M = (const float*)d_in[1];      // (D, D) fp32
  float* out = (float*)d_out;                  // (B, 3D, T) fp32

  float* mean   = (float*)d_ws;                          // B*D floats
  float* rowmax = mean + kB * kD;                        // B*T floats
  float* rowinv = rowmax + kB * kT;                      // B*T floats
  __bf16* Mt    = (__bf16*)(rowinv + kB * kT);           // D*D bf16 (512 KB)

  cci_mprep_kernel<<<dim3(16, 16), dim3(256), 0, stream>>>(M, Mt);
  cci_mean_kernel<<<dim3(kB * kD), dim3(256), 0, stream>>>(x, mean);
  cci_softmax_stats_kernel<<<dim3(kT / 256, kB), dim3(256), 0, stream>>>(
      x, out, rowmax, rowinv);
  cci_wmma_gemm_kernel<<<dim3(kT / kTileT, kB), dim3(256), kSmemK3, stream>>>(
      x, Mt, mean, rowmax, rowinv, out);
}
